// MHA_80728205296225
// MI455X (gfx1250) — compile-verified, matrix operand order checked by reading, not run on hardware
//
#include <hip/hip_runtime.h>
#include <hip/hip_bf16.h>

// ---------------------------------------------------------------------------
// MHA for gfx1250 (MI455X): bf16 WMMA with fp32 accumulation.
// bs=2, seq=2048, EMBED=1024, H=16, hd=64.
// All WMMA operands (A and B) are loaded via one "A layout" helper from
// contiguous-K storage: weights are pre-transposed to [N][K], V is stored
// transposed [bs,H,hd,seq]. Every fragment = two global_load_b128.
// ---------------------------------------------------------------------------

#define SEQ   2048
#define EMB   1024
#define HD    64
#define NH    16
#define BS    2
#define QKVN  (3 * NH * HD)   // 3072

typedef __attribute__((ext_vector_type(16))) __bf16 v16bf;
typedef __attribute__((ext_vector_type(8)))  float  v8f;

static __device__ __forceinline__ v8f wmma_bf16(v16bf a, v16bf b, v8f c) {
  // (neg_a, A, neg_b, B, c_mod, C, reuse_a, reuse_b)
  return __builtin_amdgcn_wmma_f32_16x16x32_bf16(false, a, false, b, (short)0, c,
                                                 false, false);
}

// 16x32 bf16 operand fragment from a row-major [rows][lda] matrix with
// contiguous K. Row M = lane%16; K-pair pattern per ISA:
// lanes 0-15: K={0..7,16..23}; lanes 16-31: K={8..15,24..31}.
// Per lane this is two contiguous 16-byte runs -> two b128 loads.
// Used for A operands directly, and for B operands on transposed storage
// (column-per-lane with identical K indexing).
static __device__ __forceinline__ v16bf frag_a(const __bf16* A, int lda,
                                               int row0, int k0, int lane) {
  const int half = (lane >> 4) & 1;
  const int m    = lane & 15;
  const __bf16* p = A + (size_t)(row0 + m) * lda + k0 + half * 8;
  v16bf f;
#pragma unroll
  for (int r = 0; r < 8; ++r) {
    const int kk = 2 * (r & 3) + (r >> 2) * 16;
    f[2 * r]     = p[kk];
    f[2 * r + 1] = p[kk + 1];
  }
  return f;
}

// ---------------------------------------------------------------------------
__global__ void cvt_f32_bf16(const float* __restrict__ in,
                             __bf16* __restrict__ out, int n) {
  int i = blockIdx.x * blockDim.x + threadIdx.x;
  if (i < n) out[i] = (__bf16)in[i];
}

// Transpose + convert: in [K][N] f32 row-major -> out [N][K] bf16.
// 32x32 tiles through LDS. Grid: (N/32, K/32), block (32,8).
__global__ __launch_bounds__(256) void transpose_cvt(
    const float* __restrict__ in, __bf16* __restrict__ out, int K, int N) {
  __shared__ __bf16 t[32][33];
  const int kb = blockIdx.y * 32, nb = blockIdx.x * 32;
#pragma unroll
  for (int i = threadIdx.y; i < 32; i += 8)
    t[i][threadIdx.x] = (__bf16)in[(size_t)(kb + i) * N + nb + threadIdx.x];
  __syncthreads();
#pragma unroll
  for (int i = threadIdx.y; i < 32; i += 8)
    out[(size_t)(nb + i) * K + kb + threadIdx.x] = t[threadIdx.x][i];
}

// ---------------------------------------------------------------------------
// QKV projection: [4096 x 1024] @ W^T[3072 x 1024] -> scatter into
// Q [bs,H,seq,hd], K [bs,H,seq,hd], V^T [bs,H,hd,seq] (bf16).
// Q is pre-scaled by 1/sqrt(hd). Each wave: 16x64 tile; each 64-col tile
// lives entirely inside one (h, q/k/v) group -> scalar-uniform scatter.
// Grid: x = 3072/256 = 12, y = 4096/16 = 256; block 128 (4 waves).
__global__ __launch_bounds__(128) void qkv_gemm(
    const __bf16* __restrict__ xb, const __bf16* __restrict__ wt,
    __bf16* __restrict__ qb, __bf16* __restrict__ kb, __bf16* __restrict__ vtb) {
  const int lane  = threadIdx.x & 31;
  const int wave  = __builtin_amdgcn_readfirstlane(threadIdx.x >> 5);
  const int tileM = blockIdx.y * 16;
  const int tileN = blockIdx.x * 256 + wave * 64;

  v8f acc[4] = {};
  for (int k = 0; k < EMB; k += 32) {
    v16bf a = frag_a(xb, EMB, tileM, k, lane);
#pragma unroll
    for (int t = 0; t < 4; ++t) {
      v16bf b = frag_a(wt, EMB, tileN + t * 16, k, lane);  // W^T rows = out cols
      acc[t] = wmma_bf16(a, b, acc[t]);
    }
  }

  // Wave-uniform decomposition of the 64-col group: col = (h*3+sel)*64 + d.
  const int g   = tileN >> 6;   // scalar
  const int sel = g % 3;
  const int h   = g / 3;
  const int half = (lane >> 4) & 1;
  const int n    = lane & 15;

#pragma unroll
  for (int t = 0; t < 4; ++t) {
#pragma unroll
    for (int r = 0; r < 8; ++r) {
      const int row = tileM + r + half * 8;   // token index in [0,4096)
      const int d   = t * 16 + n;             // 0..63
      const int bi  = row >> 11;
      const int s   = row & (SEQ - 1);
      const int bh  = bi * NH + h;
      const float v = acc[t][r];
      if (sel == 0)
        qb[(((size_t)bh) * SEQ + s) * HD + d] = (__bf16)(v * 0.125f);  // 1/sqrt(64)
      else if (sel == 1)
        kb[(((size_t)bh) * SEQ + s) * HD + d] = (__bf16)v;
      else
        vtb[(((size_t)bh) * HD + d) * SEQ + s] = (__bf16)v;            // transposed
    }
  }
}

// ---------------------------------------------------------------------------
// Flash attention: grid.y = bs*H (32), grid.x = seq/64 (32); 4 waves/block,
// each wave owns 16 query rows and streams KV in 32-key tiles.
__global__ __launch_bounds__(128) void attn_kernel(
    const __bf16* __restrict__ qb, const __bf16* __restrict__ kb,
    const __bf16* __restrict__ vtb, __bf16* __restrict__ ab) {
  __shared__ __bf16 Pbuf[4][16 * 32];  // per-wave P staging (C-layout -> A-layout)

  const int lane = threadIdx.x & 31;
  const int wave = threadIdx.x >> 5;
  const int half = (lane >> 4) & 1;
  const int nn   = lane & 15;
  const int bh   = blockIdx.y;
  const int q0   = blockIdx.x * 64 + wave * 16;

  const __bf16* Q  = qb  + (size_t)bh * SEQ * HD;
  const __bf16* K  = kb  + (size_t)bh * SEQ * HD;
  const __bf16* Vt = vtb + (size_t)bh * HD * SEQ;   // [hd][seq]
  __bf16* P = &Pbuf[wave][0];

  const v16bf qa0 = frag_a(Q, HD, q0, 0,  lane);   // dims 0..31
  const v16bf qa1 = frag_a(Q, HD, q0, 32, lane);   // dims 32..63

  v8f o[4] = {};
  float m[8], l[8];
#pragma unroll
  for (int r = 0; r < 8; ++r) { m[r] = -1e30f; l[r] = 0.0f; }

  for (int j = 0; j < SEQ; j += 32) {
    // Prefetch next KV tile into cache while this one computes.
    const int jn = (j + 32) & (SEQ - 1);
    __builtin_prefetch(K + (size_t)jn * HD, 0, 1);
    __builtin_prefetch(Vt + jn, 0, 1);

    // scores: Q[16x64] @ K^T -> two 16x16 tiles (keys j..j+15, j+16..j+31).
    // K rows in "A layout" == K^T in B layout.
    v16bf kb00 = frag_a(K, HD, j,      0,  lane);
    v16bf kb01 = frag_a(K, HD, j,      32, lane);
    v16bf kb10 = frag_a(K, HD, j + 16, 0,  lane);
    v16bf kb11 = frag_a(K, HD, j + 16, 32, lane);
    v8f s0 = {}, s1 = {};
    s0 = wmma_bf16(qa0, kb00, s0);
    s0 = wmma_bf16(qa1, kb01, s0);
    s1 = wmma_bf16(qa0, kb10, s1);
    s1 = wmma_bf16(qa1, kb11, s1);

    // online softmax over the 32 new keys; row = r + half*8, cols across lanes.
#pragma unroll
    for (int r = 0; r < 8; ++r) {
      float t = fmaxf(s0[r], s1[r]);
      t = fmaxf(t, __shfl_xor(t, 1, 16));
      t = fmaxf(t, __shfl_xor(t, 2, 16));
      t = fmaxf(t, __shfl_xor(t, 4, 16));
      t = fmaxf(t, __shfl_xor(t, 8, 16));
      const float mn    = fmaxf(m[r], t);
      const float alpha = __expf(m[r] - mn);
      const float p0    = __expf(s0[r] - mn);
      const float p1    = __expf(s1[r] - mn);
      float rs = p0 + p1;
      rs += __shfl_xor(rs, 1, 16);
      rs += __shfl_xor(rs, 2, 16);
      rs += __shfl_xor(rs, 4, 16);
      rs += __shfl_xor(rs, 8, 16);
      l[r] = l[r] * alpha + rs;
      m[r] = mn;
#pragma unroll
      for (int t2 = 0; t2 < 4; ++t2) o[t2][r] *= alpha;
      const int row = r + half * 8;
      P[row * 32 + nn]      = (__bf16)p0;
      P[row * 32 + 16 + nn] = (__bf16)p1;
    }
    asm volatile("s_wait_dscnt 0x0" ::: "memory");

    // O += P[16x32] @ V[32x64]; V^T rows = output dims, contiguous in keys.
    v16bf pa = frag_a(P, 32, 0, 0, lane);
#pragma unroll
    for (int t2 = 0; t2 < 4; ++t2) {
      v16bf vf = frag_a(Vt, SEQ, t2 * 16, j, lane);
      o[t2] = wmma_bf16(pa, vf, o[t2]);
    }
  }

  // normalize + store; attn buffer flat offset = (bh*seq + s)*hd + d,
  // which row-major-reinterprets exactly like the reference's .reshape.
#pragma unroll
  for (int r = 0; r < 8; ++r) {
    const float inv = 1.0f / l[r];
    const int row = q0 + r + half * 8;
#pragma unroll
    for (int t2 = 0; t2 < 4; ++t2) {
      ab[((size_t)bh * SEQ + row) * HD + t2 * 16 + nn] = (__bf16)(o[t2][r] * inv);
    }
  }
}

// ---------------------------------------------------------------------------
// Output projection: [4096 x 1024] @ Wln^T[1024 x 1024] + bias -> fp32 out.
// Grid: x = 1024/256 = 4, y = 4096/16 = 256; 4 waves/block.
__global__ __launch_bounds__(128) void out_gemm(
    const __bf16* __restrict__ ab, const __bf16* __restrict__ wlnt,
    const float* __restrict__ bias, float* __restrict__ out) {
  const int lane  = threadIdx.x & 31;
  const int wave  = __builtin_amdgcn_readfirstlane(threadIdx.x >> 5);
  const int tileM = blockIdx.y * 16;
  const int tileN = blockIdx.x * 256 + wave * 64;

  v8f acc[4] = {};
  for (int k = 0; k < EMB; k += 32) {
    v16bf a = frag_a(ab, EMB, tileM, k, lane);
#pragma unroll
    for (int t = 0; t < 4; ++t) {
      v16bf b = frag_a(wlnt, EMB, tileN + t * 16, k, lane);
      acc[t] = wmma_bf16(a, b, acc[t]);
    }
  }

  const int half = (lane >> 4) & 1;
  const int n    = lane & 15;
#pragma unroll
  for (int t = 0; t < 4; ++t) {
#pragma unroll
    for (int r = 0; r < 8; ++r) {
      const int row = tileM + r + half * 8;
      const int col = tileN + t * 16 + n;
      out[(size_t)row * EMB + col] = acc[t][r] + bias[col];
    }
  }
}

// ---------------------------------------------------------------------------
extern "C" void kernel_launch(void* const* d_in, const int* in_sizes, int n_in,
                              void* d_out, int out_size, void* d_ws, size_t ws_size,
                              hipStream_t stream) {
  const float* x    = (const float*)d_in[0];  // [2,2048,1024]
  const float* Wqkv = (const float*)d_in[1];  // [1024,3072]
  const float* Wln  = (const float*)d_in[2];  // [1024,1024]
  const float* bln  = (const float*)d_in[3];  // [1024]

  const size_t X_E   = (size_t)BS * SEQ * EMB;      // 4,194,304
  const size_t WQ_E  = (size_t)EMB * QKVN;          // 3,145,728
  const size_t WL_E  = (size_t)EMB * EMB;           // 1,048,576
  const size_t QKV_E = (size_t)BS * NH * SEQ * HD;  // 4,194,304 each

  char* ws = (char*)d_ws;
  __bf16* xb    = (__bf16*)ws;  ws += X_E   * sizeof(__bf16);
  __bf16* wqkvt = (__bf16*)ws;  ws += WQ_E  * sizeof(__bf16);  // [3072][1024]
  __bf16* wlnt  = (__bf16*)ws;  ws += WL_E  * sizeof(__bf16);  // [1024][1024] (transposed)
  __bf16* qb    = (__bf16*)ws;  ws += QKV_E * sizeof(__bf16);
  __bf16* kbuf  = (__bf16*)ws;  ws += QKV_E * sizeof(__bf16);
  __bf16* vtb   = (__bf16*)ws;  ws += QKV_E * sizeof(__bf16);  // [bs,H,hd,seq]
  __bf16* abuf  = (__bf16*)ws;  ws += X_E   * sizeof(__bf16);

  cvt_f32_bf16<<<(int)((X_E + 255) / 256), 256, 0, stream>>>(x, xb, (int)X_E);
  transpose_cvt<<<dim3(QKVN / 32, EMB / 32), dim3(32, 8), 0, stream>>>(Wqkv, wqkvt, EMB, QKVN);
  transpose_cvt<<<dim3(EMB / 32, EMB / 32), dim3(32, 8), 0, stream>>>(Wln, wlnt, EMB, EMB);

  qkv_gemm<<<dim3(QKVN / 256, (BS * SEQ) / 16), 128, 0, stream>>>(xb, wqkvt, qb, kbuf, vtb);
  attn_kernel<<<dim3(SEQ / 64, BS * NH), 128, 0, stream>>>(qb, kbuf, vtb, abuf);
  out_gemm<<<dim3(EMB / 256, (BS * SEQ) / 16), 128, 0, stream>>>(abuf, wlnt, bln, (float*)d_out);
}
